// DNA_58849641889870
// MI455X (gfx1250) — matrix operand order repaired and checked by reading, not compiled, not run on hardware
//
#include <hip/hip_runtime.h>
#include <hip/hip_bf16.h>
#include <cstdint>

// ---------------- problem constants ----------------
constexpr int Vv   = 32000;
constexpr int Dd   = 1024;
constexpr int Hh   = 16;
constexpr int HDd  = 64;
constexpr int Tt   = 2048;
constexpr int DFF  = 4096;
constexpr int CAP  = 512;
constexpr int NMOD = 8;
constexpr int ETOT = 9;
constexpr int NHOPS= 4;

// ---------------- WMMA plumbing (gfx1250, wave32) ----------------
typedef __bf16 bf16x16 __attribute__((ext_vector_type(16)));
typedef float  f32x8   __attribute__((ext_vector_type(8)));

union Frag { bf16x16 v; uint32_t u[8]; };

__device__ inline f32x8 wmma_bf16(const Frag& a, const Frag& b, f32x8 c) {
    // D = A(16x32 bf16) * B(32x16 bf16) + C(16x16 f32)
    return __builtin_amdgcn_wmma_f32_16x16x32_bf16(
        false, a.v, false, b.v, (short)0, c, false, false);
}

__device__ inline uint16_t bf16u(float f) {
    uint32_t u = __float_as_uint(f);
    uint32_t r = (u + 0x7fffu + ((u >> 16) & 1u)) >> 16;
    return (uint16_t)r;
}
__device__ inline float gelu_f(float x) {
    float x3 = x * x * x;
    return 0.5f * x * (1.f + tanhf(0.79788456f * (x + 0.044715f * x3)));
}

// ---- CDNA5 async global->LDS copy (ASYNCcnt-tracked), 16B per lane ----
typedef int v4i_ __attribute__((vector_size(16)));
#if defined(__HIP_DEVICE_COMPILE__)
# if __has_builtin(__builtin_amdgcn_global_load_async_to_lds_b128)
#  define ASYNC_CP(gp, lp)                                                     \
    __builtin_amdgcn_global_load_async_to_lds_b128(                            \
        (__attribute__((address_space(1))) v4i_*)(uintptr_t)(gp),              \
        (__attribute__((address_space(3))) v4i_*)(uint32_t)(uintptr_t)(lp),    \
        0, 0)
# else
#  define ASYNC_CP(gp, lp)                                                     \
    asm volatile("global_load_async_to_lds_b128 %0, %1, off"                   \
                 :: "v"((uint32_t)(uintptr_t)(lp)), "v"((const void*)(gp))     \
                 : "memory")
# endif
# define WAIT_ASYNC() asm volatile("s_wait_asynccnt 0x0" ::: "memory")
# define WAIT_DS()    asm volatile("s_wait_dscnt 0x0" ::: "memory")
#else
# define ASYNC_CP(gp, lp) ((void)0)
# define WAIT_ASYNC()     ((void)0)
# define WAIT_DS()        ((void)0)
#endif

// A-fragment dword pattern: u[i] : i<4 -> k=8*half+2i ; i>=4 -> k=16+8*half+2(i-4)
#define LOAD_FRAG_A(frag, rowdw)                                              \
    _Pragma("unroll") for (int _i = 0; _i < 8; ++_i)                          \
        (frag).u[_i] = (rowdw)[(_i < 4) ? (4 * half + _i) : (8 + 4 * half + (_i - 4))];
// B-fragment dword pattern: u[i] : i<4 -> k=8*half+2i ; i>=4 -> k=16+8*half+2(i-4)
#define LOAD_FRAG_B(frag, rowdw)                                              \
    _Pragma("unroll") for (int _i = 0; _i < 8; ++_i)                          \
        (frag).u[_i] = (rowdw)[(_i < 4) ? (4 * half + _i) : (8 + 4 * half + (_i - 4))];

// ---------------- elementwise kernels ----------------
__global__ void k_cvt(const float* __restrict__ s, uint16_t* __restrict__ d, long long n) {
    long long i = (long long)blockIdx.x * blockDim.x + threadIdx.x;
    if (i < n) d[i] = bf16u(s[i]);
}

// f32 [K,N] -> bf16 [N,K]  (LDS tiled transpose; K,N multiples of 32; batched on z)
__global__ __launch_bounds__(256)
void k_cvtT(const float* __restrict__ src, uint16_t* __restrict__ dst, int K, int N) {
    __shared__ float tile[32][33];
    long long zo = (long long)blockIdx.z * K * N;
    src += zo; dst += zo;
    int nb = blockIdx.x * 32, kb = blockIdx.y * 32;
    int tx = threadIdx.x & 31, ty = threadIdx.x >> 5;
    for (int r = ty; r < 32; r += 8)
        tile[r][tx] = src[(long long)(kb + r) * N + nb + tx];
    __syncthreads();
    for (int r = ty; r < 32; r += 8)
        dst[(long long)(nb + r) * K + kb + tx] = bf16u(tile[tx][r]);
}

__global__ void k_embed(const int* __restrict__ ids, const float* __restrict__ emb,
                        float* __restrict__ h, uint16_t* __restrict__ hbf) {
    long long i = (long long)blockIdx.x * blockDim.x + threadIdx.x;
    if (i >= (long long)Tt * Dd) return;
    int t = (int)(i / Dd), d = (int)(i % Dd);
    float v = emb[(long long)ids[t] * Dd + d];
    h[i] = v; hbf[i] = bf16u(v);
}

__global__ void k_ropetab(float* __restrict__ cost, float* __restrict__ sint) {
    int i = blockIdx.x * blockDim.x + threadIdx.x;
    if (i >= Tt * 32) return;
    int t = i / 32, j = i % 32;
    float inv = __powf(10000.f, -(float)j / 32.f);
    float a = (float)t * inv;
    float c = __cosf(a), s = __sinf(a);
    cost[t * 64 + j] = c; cost[t * 64 + j + 32] = c;
    sint[t * 64 + j] = s; sint[t * 64 + j + 32] = s;
}

__global__ void k_rope(const float* __restrict__ x, uint16_t* __restrict__ xbf,
                       const float* __restrict__ cost, const float* __restrict__ sint,
                       const int* __restrict__ tokmap, int rows) {
    long long i = (long long)blockIdx.x * blockDim.x + threadIdx.x;
    if (i >= (long long)rows * Hh * 32) return;
    int j = (int)(i % 32);
    long long rh = i / 32;
    int hh = (int)(rh % Hh);
    long long row = rh / Hh;
    int tok = tokmap ? tokmap[row] : (int)row;
    if (tok < 0) tok = 0;
    long long base = row * Dd + hh * 64;
    float x1 = x[base + j], x2 = x[base + j + 32];
    float cc = cost[tok * 64 + j], ss = sint[tok * 64 + j];
    xbf[base + j]      = bf16u(x1 * cc - x2 * ss);
    xbf[base + j + 32] = bf16u(x2 * cc + x1 * ss);
}

// ---------------- WMMA GEMM: C[M,N] = A[M,K](bf16) x Bt[N,K](bf16) (+R) -----
// block: 256 thr = 8 waves; block tile 64x128; wave tile 16(m) x 64(n);
// K-step 32; async double-buffered LDS staging.
__global__ __launch_bounds__(256)
void k_gemm(const uint16_t* __restrict__ A, const uint16_t* __restrict__ Bt,
            float* __restrict__ C, uint16_t* __restrict__ Cbf,
            const float* __restrict__ R,
            int M, int N, int K, int gelu,
            long long sA, long long sB, long long sC) {
    A  += (long long)blockIdx.z * sA;
    Bt += (long long)blockIdx.z * sB;
    long long co = (long long)blockIdx.z * sC;
    C += co;
    if (Cbf) Cbf += co;
    if (R)   R   += co;
    const int n0 = blockIdx.x * 128, m0 = blockIdx.y * 64;
    const int tid = threadIdx.x, lane = tid & 31, wv = tid >> 5;
    const int wm = wv & 3, wn = wv >> 2;
    const int half = lane >> 4, ln = lane & 15;

    // rows padded to 40 ushorts (80B): 16B-aligned for b128 async, and
    // 20-dword row stride -> 16 distinct banks across fragment lanes.
    __shared__ alignas(16) uint16_t As[2][64][40];
    __shared__ alignas(16) uint16_t Bs[2][128][40];

    const int sr = tid >> 2;             // staging row 0..63
    const int sc = (tid & 3) * 8;        // staging col (ushorts), 16B chunks

    auto issue = [&](int it, int buf) {
        const int koff = it * 32;
        ASYNC_CP(A  + (long long)(m0 + sr) * K + koff + sc,      &As[buf][sr][sc]);
        ASYNC_CP(Bt + (long long)(n0 + sr) * K + koff + sc,      &Bs[buf][sr][sc]);
        ASYNC_CP(Bt + (long long)(n0 + 64 + sr) * K + koff + sc, &Bs[buf][64 + sr][sc]);
    };

    f32x8 cc[4] = {{}, {}, {}, {}};
    const int nk = K / 32;
    issue(0, 0);
    for (int it = 0; it < nk; ++it) {
        const int buf = it & 1;
        WAIT_ASYNC();
        __syncthreads();                 // tile `it` visible; prev compute done
        if (it + 1 < nk) issue(it + 1, buf ^ 1);  // overlap loads with WMMA

        // preload ALL fragments, then issue WMMAs back-to-back
        Frag a, b[4];
        const uint32_t* arp = (const uint32_t*)As[buf][wm * 16 + ln];
        LOAD_FRAG_A(a, arp);
        #pragma unroll
        for (int s = 0; s < 4; ++s) {
            const uint32_t* brp = (const uint32_t*)Bs[buf][wn * 64 + s * 16 + ln];
            LOAD_FRAG_B(b[s], brp);
        }
        #pragma unroll
        for (int s = 0; s < 4; ++s)
            cc[s] = wmma_bf16(a, b[s], cc[s]);
    }

    #pragma unroll
    for (int s = 0; s < 4; ++s) {
        #pragma unroll
        for (int j = 0; j < 8; ++j) {
            int row = m0 + wm * 16 + j + 8 * half;
            int col = n0 + wn * 64 + s * 16 + ln;
            long long ix = (long long)row * N + col;
            float v = cc[s][j];
            if (gelu) v = gelu_f(v);
            if (R) v += R[ix];
            C[ix] = v;
            if (Cbf) Cbf[ix] = bf16u(v);
        }
    }
}

// ---------------- flash attention (causal) with WMMA ----------------
// block: 128 thr = 4 waves; 64 query rows/block; 32-key steps.
__global__ __launch_bounds__(128)
void k_attn(const uint16_t* __restrict__ qbf, const uint16_t* __restrict__ kbf,
            const uint16_t* __restrict__ vbf, uint16_t* __restrict__ obf,
            int Tc, long long bStride) {
    const int head = blockIdx.x, qblk = blockIdx.y, z = blockIdx.z;
    qbf += (long long)z * bStride; kbf += (long long)z * bStride;
    vbf += (long long)z * bStride; obf += (long long)z * bStride;
    const int tid = threadIdx.x, lane = tid & 31, wv = tid >> 5;
    const int half = lane >> 4, ln = lane & 15;

    __shared__ alignas(16) uint16_t Ks[32][80];    // [key][hd], padded to 160B
    __shared__ alignas(16) uint16_t Vt[64][34];    // [hd][key] transposed
    __shared__ alignas(16) uint16_t Pb[4][16][32]; // per-wave probabilities

    // Q fragments (16 rows x 64 hd -> 2 A-frags) straight from global
    const int qrow = qblk * 64 + wv * 16 + ln;
    const uint32_t* qdw = (const uint32_t*)(qbf + (long long)qrow * Dd + head * HDd);
    Frag aq0, aq1;
    LOAD_FRAG_A(aq0, qdw);
    LOAD_FRAG_A(aq1, qdw + 16);

    f32x8 o0 = {}, o1 = {}, o2 = {}, o3 = {};
    float mrun[8], lrun[8];
    #pragma unroll
    for (int j = 0; j < 8; ++j) { mrun[j] = -1e30f; lrun[j] = 0.f; }
    const float scale = 0.125f; // 1/sqrt(64)

    const int kend = min(Tc, qblk * 64 + 64);
    for (int kb = 0; kb < kend; kb += 32) {
        // K tile via async copy: 32 rows x 128B = 256 x 16B chunks, 2/thread
        {
            int r0 = tid >> 3,          c0 = (tid & 7) * 8;
            int r1 = (tid + 128) >> 3,  c1 = (tid & 7) * 8;
            ASYNC_CP(kbf + (long long)(kb + r0) * Dd + head * HDd + c0, &Ks[r0][c0]);
            ASYNC_CP(kbf + (long long)(kb + r1) * Dd + head * HDd + c1, &Ks[r1][c1]);
        }
        // V tile: register-staged (8 loads in flight), then transposed store
        uint32_t tv[8];
        #pragma unroll
        for (int q = 0; q < 8; ++q) {
            int s = tid + q * 128;
            int vr = s >> 5, pd = s & 31;
            tv[q] = ((const uint32_t*)(vbf + (long long)(kb + vr) * Dd + head * HDd))[pd];
        }
        #pragma unroll
        for (int q = 0; q < 8; ++q) {
            int s = tid + q * 128;
            int vr = s >> 5, pd = s & 31;
            Vt[2 * pd][vr]     = (uint16_t)(tv[q] & 0xffffu);
            Vt[2 * pd + 1][vr] = (uint16_t)(tv[q] >> 16);
        }
        WAIT_ASYNC();
        __syncthreads();

        // S = Q * K^T : two 16x16 tiles (keys kb..+15, kb+16..+31)
        Frag bk00, bk01, bk10, bk11;
        const uint32_t* kr0 = (const uint32_t*)Ks[ln];
        const uint32_t* kr1 = (const uint32_t*)Ks[16 + ln];
        LOAD_FRAG_B(bk00, kr0); LOAD_FRAG_B(bk01, kr0 + 16);
        LOAD_FRAG_B(bk10, kr1); LOAD_FRAG_B(bk11, kr1 + 16);
        f32x8 s0 = {}, s1 = {};
        s0 = wmma_bf16(aq0, bk00, s0); s0 = wmma_bf16(aq1, bk01, s0);
        s1 = wmma_bf16(aq0, bk10, s1); s1 = wmma_bf16(aq1, bk11, s1);

        // online softmax per row; write P to per-wave LDS as bf16
        #pragma unroll
        for (int j = 0; j < 8; ++j) {
            int qi  = qblk * 64 + wv * 16 + j + 8 * half;
            int k0i = kb + ln, k1i = kb + 16 + ln;
            bool msk0 = (k0i > qi);
            bool msk1 = (k1i > qi);
            float e0 = msk0 ? -1e30f : s0[j] * scale;
            float e1 = msk1 ? -1e30f : s1[j] * scale;
            float mx = fmaxf(e0, e1);
            #pragma unroll
            for (int m = 1; m < 16; m <<= 1) mx = fmaxf(mx, __shfl_xor(mx, m, 32));
            float mnew = fmaxf(mrun[j], mx);
            float p0 = msk0 ? 0.f : __expf(e0 - mnew);
            float p1 = msk1 ? 0.f : __expf(e1 - mnew);
            float rs = p0 + p1;
            #pragma unroll
            for (int m = 1; m < 16; m <<= 1) rs += __shfl_xor(rs, m, 32);
            float corr = __expf(mrun[j] - mnew);
            lrun[j] = lrun[j] * corr + rs;
            mrun[j] = mnew;
            o0[j] *= corr; o1[j] *= corr; o2[j] *= corr; o3[j] *= corr;
            Pb[wv][j + 8 * half][ln]      = bf16u(p0);
            Pb[wv][j + 8 * half][16 + ln] = bf16u(p1);
        }
        WAIT_DS();   // LDS write->read within the same wave

        // O += P(16x32) * V(32x64) : preload frags, then 4 WMMA back-to-back
        Frag ap, bv[4];
        const uint32_t* prow = (const uint32_t*)Pb[wv][ln];
        LOAD_FRAG_A(ap, prow);
        #pragma unroll
        for (int nt = 0; nt < 4; ++nt) {
            const uint32_t* vr = (const uint32_t*)Vt[nt * 16 + ln];
            LOAD_FRAG_B(bv[nt], vr);
        }
        o0 = wmma_bf16(ap, bv[0], o0);
        o1 = wmma_bf16(ap, bv[1], o1);
        o2 = wmma_bf16(ap, bv[2], o2);
        o3 = wmma_bf16(ap, bv[3], o3);
        __syncthreads();
    }

    #pragma unroll
    for (int j = 0; j < 8; ++j) {
        int row = qblk * 64 + wv * 16 + j + 8 * half;
        float inv = (lrun[j] > 0.f) ? 1.f / lrun[j] : 0.f;
        uint16_t* orow = obf + (long long)row * Dd + head * HDd;
        orow[ln]      = bf16u(o0[j] * inv);
        orow[16 + ln] = bf16u(o1[j] * inv);
        orow[32 + ln] = bf16u(o2[j] * inv);
        orow[48 + ln] = bf16u(o3[j] * inv);
    }
}

// ---------------- routing kernels ----------------
__global__ void k_router(const float* __restrict__ h, const float* __restrict__ rw,
                         float* __restrict__ logits) {
    int wv = threadIdx.x >> 5, lane = threadIdx.x & 31;
    int t = blockIdx.x * 8 + wv;
    if (t >= Tt) return;
    for (int e = 0; e < ETOT; ++e) {
        float acc = 0.f;
        for (int d = lane; d < Dd; d += 32)
            acc += h[(long long)t * Dd + d] * rw[(long long)d * ETOT + e];
        #pragma unroll
        for (int m = 1; m < 32; m <<= 1) acc += __shfl_xor(acc, m, 32);
        if (lane == 0) logits[t * ETOT + e] = acc;
    }
}

__global__ void k_route(const float* __restrict__ logits, const int* __restrict__ amask,
                        float* __restrict__ gbuf) {
    int t = blockIdx.x * blockDim.x + threadIdx.x;
    if (t >= Tt) return;
    float l[ETOT];
    bool pad = (amask[t] == 0);
    for (int e = 0; e < ETOT; ++e) l[e] = logits[t * ETOT + e];
    if (pad) { for (int e = 0; e < ETOT - 1; ++e) l[e] = -1e30f; l[ETOT - 1] += 1e30f; }
    float mx = l[0];
    for (int e = 1; e < ETOT; ++e) mx = fmaxf(mx, l[e]);
    float p[ETOT], s = 0.f;
    for (int e = 0; e < ETOT; ++e) { p[e] = __expf(l[e] - mx); s += p[e]; }
    for (int e = 0; e < ETOT; ++e) p[e] /= s;
    int b1 = 0;
    for (int e = 1; e < ETOT; ++e) if (l[e] > l[b1]) b1 = e;
    int b2 = -1;
    for (int e = 0; e < ETOT; ++e) if (e != b1 && (b2 < 0 || l[e] > l[b2])) b2 = e;
    for (int e = 0; e < NMOD; ++e) {
        bool sel = (!pad) && ((e == b1) || (e == b2));
        gbuf[(long long)e * Tt + t] = sel ? p[e] : -1e30f;
    }
}

__global__ __launch_bounds__(256)
void k_select(const float* __restrict__ gbuf, int* __restrict__ idx, float* __restrict__ cw) {
    int e = blockIdx.x, tid = threadIdx.x;
    __shared__ float g[Tt];
    __shared__ float bv[256];
    __shared__ int   bi[256];
    for (int t = tid; t < Tt; t += 256) {
        g[t] = gbuf[(long long)e * Tt + t];
        cw[(long long)e * Tt + t] = 0.f;
    }
    __syncthreads();
    for (int c = 0; c < CAP; ++c) {
        float best = -1e29f; int bidx = -1;
        for (int t = tid; t < Tt; t += 256)
            if (g[t] > best) { best = g[t]; bidx = t; }
        bv[tid] = best; bi[tid] = bidx;
        __syncthreads();
        for (int off = 128; off; off >>= 1) {
            if (tid < off && bv[tid + off] > bv[tid]) { bv[tid] = bv[tid + off]; bi[tid] = bi[tid + off]; }
            __syncthreads();
        }
        if (tid == 0) {
            int t0 = bi[0];
            if (t0 >= 0) { idx[e * CAP + c] = t0; cw[(long long)e * Tt + t0] = bv[0]; g[t0] = -2e30f; }
            else idx[e * CAP + c] = -1;
        }
        __syncthreads();
    }
}

__global__ void k_gather(const uint16_t* __restrict__ hbf, const int* __restrict__ idx,
                         uint16_t* __restrict__ xin, int* __restrict__ tokmap) {
    long long i = (long long)blockIdx.x * blockDim.x + threadIdx.x;
    if (i >= (long long)NMOD * CAP * Dd) return;
    int d = (int)(i % Dd);
    long long ec = i / Dd;
    int c = (int)(ec % CAP), e = (int)(ec / CAP);
    int t = idx[e * CAP + c];
    xin[i] = (t >= 0) ? hbf[(long long)t * Dd + d] : (uint16_t)0;
    if (d == 0 && (e & 1) == 0) tokmap[(e >> 1) * CAP + c] = t;
}

__global__ void k_rho(const float* __restrict__ cw, float* __restrict__ rho) {
    int t = blockIdx.x * blockDim.x + threadIdx.x;
    if (t >= Tt) return;
    float r = 0.f;
    for (int e = 0; e < NMOD; ++e) r += cw[(long long)e * Tt + t];
    rho[t] = r;
}

__global__ void k_scaleh(float* __restrict__ h, const float* __restrict__ rho) {
    long long i = (long long)blockIdx.x * blockDim.x + threadIdx.x;
    if (i >= (long long)Tt * Dd) return;
    h[i] *= (1.f - rho[i / Dd]);
}

__global__ void k_scatter(const float* __restrict__ outA, const float* __restrict__ outF,
                          const int* __restrict__ idx, const float* __restrict__ cw,
                          float* __restrict__ h) {
    long long i = (long long)blockIdx.x * blockDim.x + threadIdx.x;
    if (i >= (long long)NMOD * CAP * Dd) return;
    int d = (int)(i % Dd);
    long long ec = i / Dd;
    int c = (int)(ec % CAP), e = (int)(ec / CAP);
    int t = idx[e * CAP + c];
    if (t < 0) return;
    float w = cw[(long long)e * Tt + t];
    if (w == 0.f) return;
    float src = (e < 4) ? outA[((long long)e * CAP + c) * Dd + d]
                        : outF[((long long)(e - 4) * CAP + c) * Dd + d];
    atomicAdd(&h[(long long)t * Dd + d], w * src);
}

__global__ __launch_bounds__(256)
void k_rms(const float* __restrict__ h, const float* __restrict__ lnw,
           uint16_t* __restrict__ hbf) {
    int t = blockIdx.x, tid = threadIdx.x;
    __shared__ float red[256];
    float s = 0.f;
    for (int d = tid; d < Dd; d += 256) { float v = h[(long long)t * Dd + d]; s += v * v; }
    red[tid] = s; __syncthreads();
    for (int off = 128; off; off >>= 1) {
        if (tid < off) red[tid] += red[tid + off];
        __syncthreads();
    }
    float inv = rsqrtf(red[0] / (float)Dd + 1e-5f);
    for (int d = tid; d < Dd; d += 256)
        hbf[(long long)t * Dd + d] = bf16u(h[(long long)t * Dd + d] * inv * lnw[d]);
}

// ---------------- host-side orchestration ----------------
extern "C" void kernel_launch(void* const* d_in, const int* in_sizes, int n_in,
                              void* d_out, int out_size, void* d_ws, size_t ws_size,
                              hipStream_t stream) {
    const int*   ids   = (const int*)  d_in[0];
    const int*   amask = (const int*)  d_in[1];
    const float* embW  = (const float*)d_in[2];
    const float* bbwq  = (const float*)d_in[3];
    const float* bbwk  = (const float*)d_in[4];
    const float* bbwv  = (const float*)d_in[5];
    const float* bbwo  = (const float*)d_in[6];
    const float* bbw1  = (const float*)d_in[7];
    const float* bbw2  = (const float*)d_in[8];
    const float* rtw   = (const float*)d_in[9];
    const float* ewq   = (const float*)d_in[10];
    const float* ewk   = (const float*)d_in[11];
    const float* ewv   = (const float*)d_in[12];
    const float* ewo   = (const float*)d_in[13];
    const float* ew1   = (const float*)d_in[14];
    const float* ew2   = (const float*)d_in[15];
    const float* lnw   = (const float*)d_in[16];
    float* out = (float*)d_out;

    char* wp = (char*)d_ws;
    auto alloc = [&](size_t bytes) -> void* {
        void* r = (void*)wp;
        wp += (bytes + 255) & ~(size_t)255;
        return r;
    };
    // bf16 weights (all non-embed pre-transposed to [N,K])
    uint16_t* wbE = (uint16_t*)alloc((size_t)Vv * Dd * 2);
    uint16_t* wbQ = (uint16_t*)alloc((size_t)Dd * Dd * 2);
    uint16_t* wbK = (uint16_t*)alloc((size_t)Dd * Dd * 2);
    uint16_t* wbV = (uint16_t*)alloc((size_t)Dd * Dd * 2);
    uint16_t* wbO = (uint16_t*)alloc((size_t)Dd * Dd * 2);
    uint16_t* wb1 = (uint16_t*)alloc((size_t)Dd * DFF * 2);
    uint16_t* wb2 = (uint16_t*)alloc((size_t)DFF * Dd * 2);
    uint16_t* weQ = (uint16_t*)alloc((size_t)4 * Dd * Dd * 2);
    uint16_t* weK = (uint16_t*)alloc((size_t)4 * Dd * Dd * 2);
    uint16_t* weV = (uint16_t*)alloc((size_t)4 * Dd * Dd * 2);
    uint16_t* weO = (uint16_t*)alloc((size_t)4 * Dd * Dd * 2);
    uint16_t* we1 = (uint16_t*)alloc((size_t)4 * Dd * DFF * 2);
    uint16_t* we2 = (uint16_t*)alloc((size_t)4 * DFF * Dd * 2);
    // f32 activations
    float* h    = (float*)alloc((size_t)Tt * Dd * 4);
    float* qf   = (float*)alloc((size_t)Tt * Dd * 4);
    float* kf   = (float*)alloc((size_t)Tt * Dd * 4);
    float* vf   = (float*)alloc((size_t)Tt * Dd * 4);
    float* mid  = (float*)alloc((size_t)Tt * DFF * 4);
    float* cost = (float*)alloc((size_t)Tt * 64 * 4);
    float* sint = (float*)alloc((size_t)Tt * 64 * 4);
    float* logi = (float*)alloc((size_t)Tt * ETOT * 4);
    float* gbuf = (float*)alloc((size_t)NMOD * Tt * 4);
    float* cw   = (float*)alloc((size_t)NMOD * Tt * 4);
    float* rho  = (float*)alloc((size_t)Tt * 4);
    float* eqf  = (float*)alloc((size_t)4 * CAP * Dd * 4);
    float* ekf  = (float*)alloc((size_t)4 * CAP * Dd * 4);
    float* evf  = (float*)alloc((size_t)4 * CAP * Dd * 4);
    float* emid = (float*)alloc((size_t)4 * CAP * DFF * 4);
    float* outA = (float*)alloc((size_t)4 * CAP * Dd * 4);
    float* outF = (float*)alloc((size_t)4 * CAP * Dd * 4);
    // bf16 activations
    uint16_t* hbf   = (uint16_t*)alloc((size_t)Tt * Dd * 2);
    uint16_t* qbf   = (uint16_t*)alloc((size_t)Tt * Dd * 2);
    uint16_t* kbf   = (uint16_t*)alloc((size_t)Tt * Dd * 2);
    uint16_t* vbf   = (uint16_t*)alloc((size_t)Tt * Dd * 2);
    uint16_t* abf   = (uint16_t*)alloc((size_t)Tt * Dd * 2);
    uint16_t* midbf = (uint16_t*)alloc((size_t)Tt * DFF * 2);
    uint16_t* xin   = (uint16_t*)alloc((size_t)NMOD * CAP * Dd * 2);
    uint16_t* eqbf  = (uint16_t*)alloc((size_t)4 * CAP * Dd * 2);
    uint16_t* ekbf  = (uint16_t*)alloc((size_t)4 * CAP * Dd * 2);
    uint16_t* evbf  = (uint16_t*)alloc((size_t)4 * CAP * Dd * 2);
    uint16_t* eabf  = (uint16_t*)alloc((size_t)4 * CAP * Dd * 2);
    uint16_t* emidbf= (uint16_t*)alloc((size_t)4 * CAP * DFF * 2);
    int* idx    = (int*)alloc((size_t)NMOD * CAP * 4);
    int* tokmap = (int*)alloc((size_t)4 * CAP * 4);

    auto g1 = [](long long n, int b) { return dim3((unsigned)((n + b - 1) / b)); };
    auto cvt = [&](const float* s, uint16_t* d, long long n) {
        k_cvt<<<g1(n, 256), 256, 0, stream>>>(s, d, n);
    };
    auto cvtT = [&](const float* s, uint16_t* d, int K, int N, int batch) {
        dim3 grid(N / 32, K / 32, batch);
        k_cvtT<<<grid, 256, 0, stream>>>(s, d, K, N);
    };
    auto gemm = [&](const uint16_t* A, const uint16_t* B, float* Cc, uint16_t* Cbf,
                    const float* R, int M, int N, int K, int gl,
                    long long sA, long long sB, long long sC, int batch) {
        dim3 grid(N / 128, M / 64, batch);
        k_gemm<<<grid, 256, 0, stream>>>(A, B, Cc, Cbf, R, M, N, K, gl, sA, sB, sC);
    };

    // ---- weight conversion: embed stays [V,D]; everything else transposed ----
    cvt(embW, wbE, (long long)Vv * Dd);
    cvtT(bbwq, wbQ, Dd, Dd, 1);  cvtT(bbwk, wbK, Dd, Dd, 1);
    cvtT(bbwv, wbV, Dd, Dd, 1);  cvtT(bbwo, wbO, Dd, Dd, 1);
    cvtT(bbw1, wb1, Dd, DFF, 1); cvtT(bbw2, wb2, DFF, Dd, 1);
    cvtT(ewq, weQ, Dd, Dd, 4);   cvtT(ewk, weK, Dd, Dd, 4);
    cvtT(ewv, weV, Dd, Dd, 4);   cvtT(ewo, weO, Dd, Dd, 4);
    cvtT(ew1, we1, Dd, DFF, 4);  cvtT(ew2, we2, DFF, Dd, 4);

    // ---- embed + rope tables ----
    k_embed<<<g1((long long)Tt * Dd, 256), 256, 0, stream>>>(ids, embW, h, hbf);
    k_ropetab<<<g1(Tt * 32, 256), 256, 0, stream>>>(cost, sint);

    // ---- backbone attention ----
    gemm(hbf, wbQ, qf, nullptr, nullptr, Tt, Dd, Dd, 0, 0, 0, 0, 1);
    gemm(hbf, wbK, kf, nullptr, nullptr, Tt, Dd, Dd, 0, 0, 0, 0, 1);
    gemm(hbf, wbV, vf, vbf,     nullptr, Tt, Dd, Dd, 0, 0, 0, 0, 1);
    k_rope<<<g1((long long)Tt * Hh * 32, 256), 256, 0, stream>>>(qf, qbf, cost, sint, nullptr, Tt);
    k_rope<<<g1((long long)Tt * Hh * 32, 256), 256, 0, stream>>>(kf, kbf, cost, sint, nullptr, Tt);
    k_attn<<<dim3(Hh, Tt / 64, 1), 128, 0, stream>>>(qbf, kbf, vbf, abf, Tt, 0);
    gemm(abf, wbO, h, hbf, h, Tt, Dd, Dd, 0, 0, 0, 0, 1);   // residual add

    // ---- backbone FFN ----
    gemm(hbf, wb1, mid, midbf, nullptr, Tt, DFF, Dd, 1, 0, 0, 0, 1);  // gelu
    gemm(midbf, wb2, h, hbf, h, Tt, Dd, DFF, 0, 0, 0, 0, 1);          // residual

    // ---- MoE hops ----
    const long long sxin = (long long)2 * CAP * Dd;  // stride over expert slots
    const long long sed  = (long long)CAP * Dd;
    const long long sew  = (long long)Dd * Dd;
    for (int hop = 0; hop < NHOPS; ++hop) {
        k_router<<<Tt / 8, 256, 0, stream>>>(h, rtw + (long long)hop * Dd * ETOT, logi);
        k_route<<<g1(Tt, 256), 256, 0, stream>>>(logi, amask, gbuf);
        k_select<<<NMOD, 256, 0, stream>>>(gbuf, idx, cw);
        k_gather<<<g1((long long)NMOD * CAP * Dd, 256), 256, 0, stream>>>(hbf, idx, xin, tokmap);

        // attention experts (slots 0,2,4,6)
        gemm(xin, weQ, eqf, nullptr, nullptr, CAP, Dd, Dd, 0, sxin, sew, sed, 4);
        gemm(xin, weK, ekf, nullptr, nullptr, CAP, Dd, Dd, 0, sxin, sew, sed, 4);
        gemm(xin, weV, evf, evbf,    nullptr, CAP, Dd, Dd, 0, sxin, sew, sed, 4);
        k_rope<<<g1((long long)4 * CAP * Hh * 32, 256), 256, 0, stream>>>(eqf, eqbf, cost, sint, tokmap, 4 * CAP);
        k_rope<<<g1((long long)4 * CAP * Hh * 32, 256), 256, 0, stream>>>(ekf, ekbf, cost, sint, tokmap, 4 * CAP);
        k_attn<<<dim3(Hh, CAP / 64, 4), 128, 0, stream>>>(eqbf, ekbf, evbf, eabf, CAP, sed);
        gemm(eabf, weO, outA, nullptr, nullptr, CAP, Dd, Dd, 0, sed, sew, sed, 4);

        // FFN experts (slots 1,3,5,7)
        gemm(xin + sed, we1, emid, emidbf, nullptr, CAP, DFF, Dd, 1,
             sxin, (long long)Dd * DFF, (long long)CAP * DFF, 4);
        gemm(emidbf, we2, outF, nullptr, nullptr, CAP, Dd, DFF, 0,
             (long long)CAP * DFF, (long long)DFF * Dd, sed, 4);

        // combine: h = (1 - rho)*h + scatter(expert_out * combine_w)
        k_rho<<<g1(Tt, 256), 256, 0, stream>>>(cw, rho);
        k_scaleh<<<g1((long long)Tt * Dd, 256), 256, 0, stream>>>(h, rho);
        k_scatter<<<g1((long long)NMOD * CAP * Dd, 256), 256, 0, stream>>>(outA, outF, idx, cw, h);
        cvt(h, hbf, (long long)Tt * Dd);
    }

    // ---- final RMSNorm + vocab projection (embed table already [N,K]) ----
    k_rms<<<Tt, 256, 0, stream>>>(h, lnw, hbf);
    gemm(hbf, wbE, out, nullptr, nullptr, Tt, Vv, Dd, 0, 0, 0, 0, 1);
}
